// GCN_10977936409091
// MI455X (gfx1250) — matrix operand order repaired and checked by reading, not compiled, not run on hardware
//
#include <hip/hip_runtime.h>
#include <hip/hip_bf16.h>
#include <stddef.h>

// ---------------------------------------------------------------------------
// Types for CDNA5 WMMA (wave32): v_wmma_f32_16x16x32_bf16
// ---------------------------------------------------------------------------
typedef __attribute__((ext_vector_type(16))) __bf16 v16bf;
typedef __attribute__((ext_vector_type(8)))  float  v8f;

#define FEATS 128   // K of both GEMMs and feature width of both aggregations

// ---------------------------------------------------------------------------
// Utility: zero a float buffer (avoids memset nodes; graph-capture safe)
// ---------------------------------------------------------------------------
__global__ void zero_kernel(float* __restrict__ p, size_t n) {
  size_t i = (size_t)blockIdx.x * blockDim.x + threadIdx.x;
  if (i < n) p[i] = 0.0f;
}

// ---------------------------------------------------------------------------
// Degrees: deg_out[src] += 1, deg_in[dst] += 1  (f32 atomics, exact < 2^24)
// ---------------------------------------------------------------------------
__global__ void degree_kernel(const int* __restrict__ src, const int* __restrict__ dst,
                              float* __restrict__ deg_out, float* __restrict__ deg_in,
                              int n_edges) {
  int e = blockIdx.x * blockDim.x + threadIdx.x;
  if (e >= n_edges) return;
  atomicAdd(&deg_out[src[e]], 1.0f);
  atomicAdd(&deg_in[dst[e]], 1.0f);
}

// norm = rsqrt(max(deg, 1))
__global__ void norm_kernel(const float* __restrict__ deg, float* __restrict__ nrm, int n) {
  int i = blockIdx.x * blockDim.x + threadIdx.x;
  if (i < n) nrm[i] = rsqrtf(fmaxf(deg[i], 1.0f));
}

// ---------------------------------------------------------------------------
// Prescale: Xs[i,:] = X[i,:] * norm_src[i]   (float4 per thread)
// ---------------------------------------------------------------------------
__global__ void prescale_kernel(const float* __restrict__ X, const float* __restrict__ nsrc,
                                float* __restrict__ Xs, int n_nodes) {
  int idx = blockIdx.x * blockDim.x + threadIdx.x;     // one float4
  int total = n_nodes * (FEATS / 4);
  if (idx >= total) return;
  int row = idx / (FEATS / 4);
  float s = nsrc[row];
  float4 v = ((const float4*)X)[idx];
  v.x *= s; v.y *= s; v.z *= s; v.w *= s;
  ((float4*)Xs)[idx] = v;
}

// ---------------------------------------------------------------------------
// Repack weights W[K x N] (row major f32) into the CDNA5 WMMA bf16 B-operand
// layout, split into hi/lo bf16 so 3 WMMAs recover ~fp32 accuracy.
// Packed index: (((t*NT + n)*32 + lane)*16 + j), t = K/32 chunk, n = N/16 tile.
//   lane<16 : element j = B[t*32 + (j&7) + (j>>3)*16    ][n*16 + lane]
//   lane>=16: element j = B[t*32 + 8 + (j&7) + (j>>3)*16][n*16 + lane-16]
// ---------------------------------------------------------------------------
__global__ void repack_w_kernel(const float* __restrict__ W, __bf16* __restrict__ Whi,
                                __bf16* __restrict__ Wlo, int N, int total /* == K*N */) {
  int idx = blockIdx.x * blockDim.x + threadIdx.x;
  if (idx >= total) return;
  const int NT = N >> 4;
  int j = idx & 15;
  int l = (idx >> 4) & 31;
  int n = (idx >> 9) % NT;
  int t = (idx >> 9) / NT;
  int row = t * 32 + (l >> 4) * 8 + (j & 7) + (j >> 3) * 16;
  int col = n * 16 + (l & 15);
  float w = W[row * N + col];
  __bf16 h = (__bf16)w;
  Whi[idx] = h;
  Wlo[idx] = (__bf16)(w - (float)h);
}

// ---------------------------------------------------------------------------
// Edge aggregation: Agg[dst[e], :] += Xs[src[e], :]
// One edge per wave32; each lane moves a float4 and issues 4 f32 atomics.
// Agg (51.2 MB) is L2-resident on MI455X (192 MB L2), so atomics stay on-chip.
// ---------------------------------------------------------------------------
__global__ void aggregate_kernel(const float* __restrict__ Xs, const int* __restrict__ src,
                                 const int* __restrict__ dst, float* __restrict__ Agg,
                                 int n_edges) {
  int gid  = blockIdx.x * blockDim.x + threadIdx.x;
  int e    = gid >> 5;
  int lane = gid & 31;
  if (e >= n_edges) return;
  int s = src[e], d = dst[e];
  float4 v = *(const float4*)(Xs + (size_t)s * FEATS + lane * 4);
  float* o = Agg + (size_t)d * FEATS + lane * 4;
  atomicAdd(o + 0, v.x);
  atomicAdd(o + 1, v.y);
  atomicAdd(o + 2, v.z);
  atomicAdd(o + 3, v.w);
}

// ---------------------------------------------------------------------------
// WMMA GEMM:  Out = f( (A * norm_dst[:,None]) @ W + bias )
//   A: [M x 128] f32, W packed hi/lo bf16, Out: [M x NT*16] f32.
//   RELU_SCALE: Out = relu(.) * norm_src[row]  (feeds next layer pre-scaled)
// Each wave owns a 16-row strip and all NT column tiles; K = 4 chunks of 32.
// 3 WMMAs per (chunk, tile): ah*bh + al*bh + ah*bl  (bf16 hi/lo split).
// ---------------------------------------------------------------------------
template <int NT, bool RELU_SCALE>
__global__ __launch_bounds__(128) void gemm_kernel(
    const float* __restrict__ A, const float* __restrict__ norm_dst,
    const float* __restrict__ norm_src, const __bf16* __restrict__ Bhi,
    const __bf16* __restrict__ Blo, const float* __restrict__ bias,
    float* __restrict__ Out, int M, int m_tiles) {
  const int wave  = threadIdx.x >> 5;
  const int strip = blockIdx.x * 4 + wave;          // 16-row strip id
  if (strip >= m_tiles) return;                     // wave-uniform exit
  const int lane = threadIdx.x & 31;
  const int half = lane >> 4;                       // 0: lanes 0-15, 1: 16-31
  const int m0   = strip * 16;

  int mrow = m0 + (lane & 15);
  if (mrow >= M) mrow = M - 1;
  const float nd = norm_dst[mrow];
  const float* arow = A + (size_t)mrow * FEATS;

  v8f acc[NT];
#pragma unroll
  for (int n = 0; n < NT; ++n) acc[n] = (v8f){0.f, 0.f, 0.f, 0.f, 0.f, 0.f, 0.f, 0.f};

#pragma unroll
  for (int t = 0; t < 4; ++t) {
    // A operand per the ISA 16-bit 16x32 layout:
    //   lane<16 holds K = t*32 + {0..7, 16..23}; lane>=16: +8.
    const int bk = t * 32 + half * 8;
    float4 f0 = *(const float4*)(arow + bk);
    float4 f1 = *(const float4*)(arow + bk + 4);
    float4 f2 = *(const float4*)(arow + bk + 16);
    float4 f3 = *(const float4*)(arow + bk + 20);
    float av[16] = {f0.x, f0.y, f0.z, f0.w, f1.x, f1.y, f1.z, f1.w,
                    f2.x, f2.y, f2.z, f2.w, f3.x, f3.y, f3.z, f3.w};
    v16bf ahi, alo;
#pragma unroll
    for (int j = 0; j < 16; ++j) {
      float x = av[j] * nd;
      __bf16 h = (__bf16)x;
      ahi[j] = h;
      alo[j] = (__bf16)(x - (float)h);
    }
#pragma unroll
    for (int n = 0; n < NT; ++n) {
      const size_t boff = (((size_t)t * NT + n) * 32 + lane) * 16;
      v16bf bhi = *(const v16bf*)(Bhi + boff);
      v16bf blo = *(const v16bf*)(Blo + boff);
      acc[n] = __builtin_amdgcn_wmma_f32_16x16x32_bf16(false, ahi, false, bhi,
                                                       (short)0, acc[n], false, false);
      acc[n] = __builtin_amdgcn_wmma_f32_16x16x32_bf16(false, alo, false, bhi,
                                                       (short)0, acc[n], false, false);
      acc[n] = __builtin_amdgcn_wmma_f32_16x16x32_bf16(false, ahi, false, blo,
                                                       (short)0, acc[n], false, false);
    }
  }

  // Epilogue: C/D layout -> row = m0 + half*8 + r, col = n*16 + lane%16
  float ns[8];
#pragma unroll
  for (int r = 0; r < 8; ++r) {
    int row = m0 + half * 8 + r;
    if (row >= M) row = M - 1;
    ns[r] = RELU_SCALE ? norm_src[row] : 1.0f;
  }
#pragma unroll
  for (int n = 0; n < NT; ++n) {
    const int c  = n * 16 + (lane & 15);
    const float bv = bias[c];
#pragma unroll
    for (int r = 0; r < 8; ++r) {
      const int row = m0 + half * 8 + r;
      float v = acc[n][r] + bv;
      if (RELU_SCALE) v = fmaxf(v, 0.f) * ns[r];
      if (row < M) Out[(size_t)row * (NT * 16) + c] = v;
    }
  }
}

// ---------------------------------------------------------------------------
// Host-side launcher
// ---------------------------------------------------------------------------
extern "C" void kernel_launch(void* const* d_in, const int* in_sizes, int n_in,
                              void* d_out, int out_size, void* d_ws, size_t ws_size,
                              hipStream_t stream) {
  const float* x   = (const float*)d_in[0];   // [N, 128]
  const int*   src = (const int*)d_in[1];     // [E]
  const int*   dst = (const int*)d_in[2];     // [E]
  const float* W1  = (const float*)d_in[3];   // [128, 128]
  const float* b1  = (const float*)d_in[4];   // [128]
  const float* W2  = (const float*)d_in[5];   // [128, 64]
  const float* b2  = (const float*)d_in[6];   // [64]

  const int n_nodes = in_sizes[0] / FEATS;    // 100000
  const int n_edges = in_sizes[1];            // 1600000
  const int ncls    = in_sizes[6];            // 64 (NT2*16 == 64)

  // Workspace carve-up (256B aligned regions)
  char* ws = (char*)d_ws;
  size_t off = 0;
  auto carve = [&](size_t bytes) -> char* {
    char* p = ws + off;
    off = (off + bytes + 255) & ~(size_t)255;
    return p;
  };
  float*  deg_out  = (float*)carve((size_t)n_nodes * 4);
  float*  deg_in   = (float*)carve((size_t)n_nodes * 4);
  float*  norm_src = (float*)carve((size_t)n_nodes * 4);
  float*  norm_dst = (float*)carve((size_t)n_nodes * 4);
  float*  xs       = (float*)carve((size_t)n_nodes * FEATS * 4);  // reused as hs
  float*  agg      = (float*)carve((size_t)n_nodes * FEATS * 4);
  __bf16* w1hi     = (__bf16*)carve((size_t)FEATS * FEATS * 2);
  __bf16* w1lo     = (__bf16*)carve((size_t)FEATS * FEATS * 2);
  __bf16* w2hi     = (__bf16*)carve((size_t)FEATS * ncls * 2);
  __bf16* w2lo     = (__bf16*)carve((size_t)FEATS * ncls * 2);

  const size_t nfeat  = (size_t)n_nodes * FEATS;
  const int    m_tiles = (n_nodes + 15) / 16;

  auto cdiv = [](size_t a, size_t b) { return (int)((a + b - 1) / b); };

  // 1) degrees + norms
  zero_kernel<<<cdiv(n_nodes, 256), 256, 0, stream>>>(deg_out, n_nodes);
  zero_kernel<<<cdiv(n_nodes, 256), 256, 0, stream>>>(deg_in, n_nodes);
  degree_kernel<<<cdiv(n_edges, 256), 256, 0, stream>>>(src, dst, deg_out, deg_in, n_edges);
  norm_kernel<<<cdiv(n_nodes, 256), 256, 0, stream>>>(deg_out, norm_src, n_nodes);
  norm_kernel<<<cdiv(n_nodes, 256), 256, 0, stream>>>(deg_in, norm_dst, n_nodes);

  // 2) weight repack (tiny; done every call for determinism)
  repack_w_kernel<<<cdiv((size_t)FEATS * FEATS, 256), 256, 0, stream>>>(
      W1, w1hi, w1lo, FEATS, FEATS * FEATS);
  repack_w_kernel<<<cdiv((size_t)FEATS * ncls, 256), 256, 0, stream>>>(
      W2, w2hi, w2lo, ncls, FEATS * ncls);

  // 3) layer 1: prescale -> aggregate -> WMMA GEMM (+bias, relu, *norm_src)
  prescale_kernel<<<cdiv(nfeat / 4, 256), 256, 0, stream>>>(x, norm_src, xs, n_nodes);
  zero_kernel<<<cdiv(nfeat, 256), 256, 0, stream>>>(agg, nfeat);
  aggregate_kernel<<<cdiv((size_t)n_edges * 32, 256), 256, 0, stream>>>(
      xs, src, dst, agg, n_edges);
  gemm_kernel<8, true><<<cdiv(m_tiles, 4), 128, 0, stream>>>(
      agg, norm_dst, norm_src, w1hi, w1lo, b1, /*Out=*/xs, n_nodes, m_tiles);

  // 4) layer 2: aggregate (xs already pre-scaled by norm_src) -> GEMM (+bias)
  zero_kernel<<<cdiv(nfeat, 256), 256, 0, stream>>>(agg, nfeat);
  aggregate_kernel<<<cdiv((size_t)n_edges * 32, 256), 256, 0, stream>>>(
      xs, src, dst, agg, n_edges);
  gemm_kernel<4, false><<<cdiv(m_tiles, 4), 128, 0, stream>>>(
      agg, norm_dst, norm_src, w2hi, w2lo, b2, (float*)d_out, n_nodes, m_tiles);

  (void)n_in; (void)out_size; (void)ws_size;
}
